// NARM_82695300317813
// MI455X (gfx1250) — matrix operand order repaired and checked by reading, not compiled, not run on hardware
//
#include <hip/hip_runtime.h>
#include <hip/hip_bf16.h>
#include <math.h>
#include <stdint.h>

typedef __attribute__((ext_vector_type(16))) __bf16 v16bf;
typedef __attribute__((ext_vector_type(8)))  float  v8f;
typedef __attribute__((ext_vector_type(2)))  float  v2f;
typedef __attribute__((ext_vector_type(4)))  unsigned int u32x4;
typedef __attribute__((ext_vector_type(8)))  int    i32x8;
typedef __attribute__((ext_vector_type(4)))  int    i32x4;

#define Tn 384
#define Bn 8
#define Dn 128
#define Hn 128
#define ROWS (Tn*Bn)   // 3072

#if defined(__has_builtin)
#  if __has_builtin(__builtin_amdgcn_tensor_load_to_lds) && __has_builtin(__builtin_amdgcn_s_wait_tensorcnt)
#    define HAVE_TDM 1
#  endif
#endif

__device__ __forceinline__ float sigm(float x)   { return 1.0f / (1.0f + __expf(-x)); }
__device__ __forceinline__ float leaky3(float x) { return x > 0.0f ? x : 0.3f * x; }

#ifdef HAVE_TDM
// 2D tile DMA: global (rows x rowlen f32, row stride in elements) -> LDS (packed)
__device__ __forceinline__ void tdm_load_2d(unsigned int lds_off, const void* gptr,
                                            unsigned int rows, unsigned int rowlen,
                                            unsigned int row_stride)
{
    unsigned long long ga = (unsigned long long)(uintptr_t)gptr;
    u32x4 g0;
    g0[0] = 1u;                                              // count=1, user mode
    g0[1] = lds_off;                                         // lds_addr
    g0[2] = (unsigned int)(ga & 0xffffffffu);                // global_addr[31:0]
    g0[3] = (unsigned int)((ga >> 32) & 0x1ffffffu) | (2u << 30); // addr[56:32] | type=2
    i32x8 g1;
    g1[0] = (int)(2u << 16);          // workgroup_mask=0, data_size=2 (4B)
    g1[1] = (int)(rowlen << 16);      // tensor_dim0[15:0] in bits 63:48
    g1[2] = (int)(rows << 16);        // tensor_dim0 hi=0, tensor_dim1[15:0] in bits 95:80
    g1[3] = (int)(rowlen << 16);      // tensor_dim1 hi=0, tile_dim0 in bits 127:112
    g1[4] = (int)rows;                // tile_dim1 (bits 143:128), tile_dim2=0
    g1[5] = (int)row_stride;          // tensor_dim0_stride[31:0]
    g1[6] = 0;                        // stride hi / dim1_stride lo
    g1[7] = 0;
    i32x4 z4 = {0, 0, 0, 0};
#if __clang_major__ >= 23
    i32x8 z8 = {0, 0, 0, 0, 0, 0, 0, 0};
    __builtin_amdgcn_tensor_load_to_lds(g0, g1, z4, z4, z8, 0);
#else
    __builtin_amdgcn_tensor_load_to_lds(g0, g1, z4, z4, 0);
#endif
}
#endif

// ---------------------------------------------------------------------------
// Weight pre-convert: W[K,N] f32  ->  Wt[N,K] bf16 (transposed)
// ---------------------------------------------------------------------------
__global__ void convert_wT_kernel(const float* __restrict__ W, __bf16* __restrict__ Wt,
                                  int K, int N)
{
    int idx = blockIdx.x * 256 + threadIdx.x;
    if (idx >= K * N) return;
    int n = idx / K, k = idx - n * K;
    Wt[(size_t)n * K + k] = (__bf16)W[(size_t)k * N + n];
}

// ---------------------------------------------------------------------------
// Generic GEMM: C[3072, N] = act( A[3072,K] @ W[K,N] + bias[N] )
// W supplied pre-transposed bf16 (Wt[N,K]) -> B fragment = one 32B load.
// bf16 WMMA 16x16x32, one 16x16 tile per wave, 8 waves / block.
// ---------------------------------------------------------------------------
__global__ void gemm_bf16_kernel(const float* __restrict__ A, int lda,
                                 const __bf16* __restrict__ Wt,
                                 const float* __restrict__ bias,
                                 float* __restrict__ C, int ldc, int coff,
                                 int K, int Ntiles, int act)
{
    int wid  = threadIdx.x >> 5;
    int lane = threadIdx.x & 31;
    int hlf  = lane >> 4;
    int mr   = lane & 15;
    int tile = blockIdx.x * 8 + wid;
    int mt = tile / Ntiles, nt = tile % Ntiles;
    int m0 = mt * 16, n0 = nt * 16;

    const __bf16* wrow = Wt + (size_t)(n0 + mr) * K + hlf * 16;
    const float*  arow0 = A + (size_t)(m0 + mr) * lda + hlf * 16;

    v8f acc = {};
    for (int kc = 0; kc < K; kc += 32) {
        // A tile (16x32 bf16 layout), built from 4 x float4 contiguous loads
        const float* arow = arow0 + kc;
        float4 a0 = *(const float4*)(arow);
        float4 a1 = *(const float4*)(arow + 4);
        float4 a2 = *(const float4*)(arow + 8);
        float4 a3 = *(const float4*)(arow + 12);
        v16bf av;
        av[0]=(__bf16)a0.x; av[1]=(__bf16)a0.y; av[2]=(__bf16)a0.z; av[3]=(__bf16)a0.w;
        av[4]=(__bf16)a1.x; av[5]=(__bf16)a1.y; av[6]=(__bf16)a1.z; av[7]=(__bf16)a1.w;
        av[8]=(__bf16)a2.x; av[9]=(__bf16)a2.y; av[10]=(__bf16)a2.z; av[11]=(__bf16)a2.w;
        av[12]=(__bf16)a3.x; av[13]=(__bf16)a3.y; av[14]=(__bf16)a3.z; av[15]=(__bf16)a3.w;

        // B tile: lane = column n, 16 consecutive K values, contiguous bf16
        v16bf bv = *(const v16bf*)(wrow + kc);

        acc = __builtin_amdgcn_wmma_f32_16x16x32_bf16(false, av, false, bv,
                                                      (short)0, acc, false, false);
    }

    int n = n0 + mr;
    float bval = bias ? bias[n] : 0.0f;
    #pragma unroll
    for (int r = 0; r < 8; r++) {
        int m = m0 + r + 8 * hlf;
        float v = acc[r] + bval;
        if (act) v = leaky3(v);
        C[(size_t)m * ldc + coff + n] = v;
    }
}

// ---------------------------------------------------------------------------
// GRU recurrence. 2 blocks (one per GRU), 384 threads = 12 waves.
// Wrt ([3H,H] bf16, transposed) held register-resident: 2 N-tiles x 4 K-chunks
// of B per wave. Each step: h -> A-format LDS tile, 8 WMMAs/wave, gates, update.
// ---------------------------------------------------------------------------
__global__ void gru_kernel(const float* __restrict__ xz0, const float* __restrict__ xz1,
                           const __bf16* __restrict__ Wrt0, const __bf16* __restrict__ Wrt1,
                           const float* __restrict__ br0, const float* __restrict__ br1,
                           float* __restrict__ out0, float* __restrict__ out1,
                           int ostr0, int ostr1)
{
    int g = blockIdx.x;
    const float*  xz  = g ? xz1  : xz0;
    const __bf16* Wrt = g ? Wrt1 : Wrt0;
    const float*  br  = g ? br1  : br0;
    float* out = g ? out1 : out0;
    int ostr   = g ? ostr1 : ostr0;

    __shared__ v16bf a_lds[4][32];        // A-format h (16x32 bf16 per K-chunk)
    __shared__ v8f   hz_lds[3 * Hn];      // hz, column-major [col][row]
    __shared__ float h_lds[Bn][Hn];

    int tid  = threadIdx.x;
    int wid  = tid >> 5;
    int lane = tid & 31;
    int hlf  = lane >> 4;
    int mr   = lane & 15;

    for (int i = tid; i < Bn * Hn; i += 384) ((float*)h_lds)[i] = 0.0f;
    if (tid < 128) { v16bf z = {}; a_lds[tid >> 5][tid & 31] = z; }

    // Preload Wrt as register-resident bf16 B tiles (contiguous 32B loads)
    v16bf Breg[2][4];
    #pragma unroll
    for (int tI = 0; tI < 2; tI++) {
        int nt = wid * 2 + tI;
        #pragma unroll
        for (int q = 0; q < 4; q++)
            Breg[tI][q] = *(const v16bf*)(Wrt + (size_t)(nt * 16 + mr) * Hn + q * 32 + hlf * 16);
    }
    __syncthreads();

    for (int t = 0; t < Tn; t++) {
        // hz = h @ Wr   (rows 0..7 valid)
        #pragma unroll
        for (int tI = 0; tI < 2; tI++) {
            v8f acc = {};
            #pragma unroll
            for (int q = 0; q < 4; q++) {
                v16bf av = a_lds[q][lane];
                acc = __builtin_amdgcn_wmma_f32_16x16x32_bf16(false, av, false, Breg[tI][q],
                                                              (short)0, acc, false, false);
            }
            int col = (wid * 2 + tI) * 16 + mr;
            if (hlf == 0) hz_lds[col] = acc;   // lanes 0-15 carry rows 0..7
        }
        __syncthreads();

        // gates: 1024 (b,c) elements over 384 threads
        const float* xzt = xz + (size_t)t * Bn * 384;
        for (int idx = tid; idx < Bn * Hn; idx += 384) {
            int b = idx >> 7, c = idx & 127;
            const float* xrow = xzt + b * 384;
            const float* hz0 = (const float*)&hz_lds[c];
            const float* hz1 = (const float*)&hz_lds[Hn + c];
            const float* hz2 = (const float*)&hz_lds[2 * Hn + c];
            float zg = sigm (xrow[c]          + hz0[b] + br[c]);
            float rg = sigm (xrow[Hn + c]     + hz1[b] + br[Hn + c]);
            float hh = tanhf(xrow[2 * Hn + c] + rg * (hz2[b] + br[2 * Hn + c]));
            float hp = h_lds[b][c];
            float hn = zg * hp + (1.0f - zg) * hh;
            h_lds[b][c] = hn;
            out[(size_t)(t * Bn + b) * ostr + c] = hn;
            // scatter into A-format bf16 tile for next step
            int q   = c >> 5;
            int kin = c & 31;
            int hi  = (kin >= 16) ? 1 : 0;
            int rem = kin & 15;
            int hf  = rem >> 3;
            int wi  = rem & 7;
            ((__bf16*)&a_lds[q][b + 16 * hi])[rem] = (__bf16)hn;
        }
        __syncthreads();
    }
}

// ---------------------------------------------------------------------------
// Causal additive attention. Grid = 8 batches x 24 i-tiles, 256 threads.
// j-tile staging of h1 / h_l via TDM (tensor_load_to_lds) when available.
// Scores in VALU; context accumulation via v_wmma_f32_16x16x4_f32.
// ---------------------------------------------------------------------------
__global__ void attn_kernel(const float* __restrict__ h1, const float* __restrict__ h2,
                            const float* __restrict__ hl, const float* __restrict__ vvec,
                            float* __restrict__ ct)
{
    int blk = blockIdx.x;
    int b  = blk / 24;
    int it = blk % 24;
    int i0 = it * 16;
    int tid  = threadIdx.x;
    int wid  = tid >> 5;
    int lane = tid & 31;
    int hlf  = lane >> 4;
    int mr   = lane & 15;

    __shared__ float h2s[16][Hn];
    __shared__ float h1s[16][Hn];
    __shared__ float hls[16][Hn];
    __shared__ float vs[Hn];
    __shared__ float S[16][16];

    for (int idx = tid; idx < 16 * Hn; idx += 256) {
        int r = idx >> 7, h = idx & 127;
        h2s[r][h] = h2[(size_t)((i0 + r) * Bn + b) * Hn + h];
    }
    if (tid < Hn) vs[tid] = vvec[tid];
    __syncthreads();

    v8f acc = {};
    int il = tid >> 4, jl = tid & 15;
    for (int jt = 0; jt <= it; jt++) {
        int j0 = jt * 16;
#ifdef HAVE_TDM
        if (wid == 0) {
            const float* g1p = h1 + (size_t)(j0 * Bn + b) * Hn;
            const float* glp = hl + (size_t)(j0 * Bn + b) * Hn;
            tdm_load_2d((unsigned int)(uintptr_t)&h1s[0][0], g1p, 16, Hn, Bn * Hn);
            tdm_load_2d((unsigned int)(uintptr_t)&hls[0][0], glp, 16, Hn, Bn * Hn);
            __builtin_amdgcn_s_wait_tensorcnt(0);
        }
        __syncthreads();
#else
        for (int idx = tid; idx < 16 * Hn; idx += 256) {
            int r = idx >> 7, h = idx & 127;
            size_t base = (size_t)((j0 + r) * Bn + b) * Hn + h;
            h1s[r][h] = h1[base];
            hls[r][h] = hl[base];
        }
        __syncthreads();
#endif

        // score(i,j) = v . sigmoid(h1[j] + h2[i]),  causal mask on diagonal tile
        float s = 0.0f;
        for (int h = 0; h < Hn; h++)
            s += vs[h] * sigm(h1s[jl][h] + h2s[il][h]);
        if (jt == it && jl > il) s = 0.0f;
        S[il][jl] = s;
        __syncthreads();

        // c_l[i, wid*16 .. +16) += S @ h_l   via fp32 WMMA 16x16x4
        #pragma unroll
        for (int kc = 0; kc < 4; kc++) {
            v2f a, bb;
            a[0]  = S[mr][kc * 4 + 2 * hlf + 0];
            a[1]  = S[mr][kc * 4 + 2 * hlf + 1];
            bb[0] = hls[kc * 4 + 2 * hlf + 0][wid * 16 + mr];
            bb[1] = hls[kc * 4 + 2 * hlf + 1][wid * 16 + mr];
            acc = __builtin_amdgcn_wmma_f32_16x16x4_f32(false, a, false, bb,
                                                        (short)0, acc, false, false);
        }
        __syncthreads();
    }

    #pragma unroll
    for (int r = 0; r < 8; r++) {
        int m = r + 8 * hlf;
        int hcol = wid * 16 + mr;
        ct[(size_t)((i0 + m) * Bn + b) * 256 + 128 + hcol] = acc[r];
    }
}

// ---------------------------------------------------------------------------
// Head finisher: pred[i] = sigmoid( sum_h A[i,h]*B[i,h] ), optional * mulsrc
// ---------------------------------------------------------------------------
__global__ void head_dot_kernel(const float* __restrict__ A, const float* __restrict__ Bv,
                                float* __restrict__ out, const float* __restrict__ mulsrc)
{
    int i = blockIdx.x * blockDim.x + threadIdx.x;
    if (i >= ROWS) return;
    float s = 0.0f;
    for (int h = 0; h < Hn; h++) s += A[(size_t)i * Hn + h] * Bv[(size_t)i * Hn + h];
    float p = sigm(s);
    if (mulsrc) p *= mulsrc[i];
    out[i] = p;
}

// ---------------------------------------------------------------------------
extern "C" void kernel_launch(void* const* d_in, const int* in_sizes, int n_in,
                              void* d_out, int out_size, void* d_ws, size_t ws_size,
                              hipStream_t stream)
{
    const float* x    = (const float*)d_in[0];
    const float* gg_k = (const float*)d_in[1];
    const float* gg_r = (const float*)d_in[2];
    const float* gg_b = (const float*)d_in[3];
    const float* gl_k = (const float*)d_in[4];
    const float* gl_r = (const float*)d_in[5];
    const float* gl_b = (const float*)d_in[6];
    const float* a1w  = (const float*)d_in[7];
    const float* a2w  = (const float*)d_in[8];
    const float* vv   = (const float*)d_in[9];
    const float* pc_w0 = (const float*)d_in[10];
    const float* pc_b0 = (const float*)d_in[11];
    const float* pc_w1 = (const float*)d_in[12];
    const float* pc_b1 = (const float*)d_in[13];
    const float* pc_v0 = (const float*)d_in[14];
    const float* pc_c0 = (const float*)d_in[15];
    const float* pc_v1 = (const float*)d_in[16];
    const float* pc_c1 = (const float*)d_in[17];
    const float* pv_w0 = (const float*)d_in[18];
    const float* pv_b0 = (const float*)d_in[19];
    const float* pv_w1 = (const float*)d_in[20];
    const float* pv_b1 = (const float*)d_in[21];
    const float* pv_v0 = (const float*)d_in[22];
    const float* pv_c0 = (const float*)d_in[23];
    const float* pv_v1 = (const float*)d_in[24];
    const float* pv_c1 = (const float*)d_in[25];
    float* out = (float*)d_out;

    // ---- float workspace ----
    float* ws   = (float*)d_ws;
    float* xz_g = ws;                         // 3072*384
    float* xz_l = xz_g + (size_t)ROWS * 384;  // 3072*384
    float* h_l  = xz_l + (size_t)ROWS * 384;  // 3072*128
    float* c_t  = h_l  + (size_t)ROWS * 128;  // 3072*256 (h_g | c_l)
    float* h1   = c_t  + (size_t)ROWS * 256;  // 3072*128
    float* h2   = h1   + (size_t)ROWS * 128;  // 3072*128
    float* a0   = h2   + (size_t)ROWS * 128;  // 3072*256
    float* a1b  = a0   + (size_t)ROWS * 256;  // 3072*128
    float* bb0  = a1b  + (size_t)ROWS * 128;  // 3072*256
    float* bb1  = bb0  + (size_t)ROWS * 256;  // 3072*128
    float* fend = bb1  + (size_t)ROWS * 128;

    // ---- bf16 transposed-weight workspace ----
    __bf16* bw = (__bf16*)fend;
    __bf16* wt_ggk = bw;                     bw += 384 * 128;
    __bf16* wt_glk = bw;                     bw += 384 * 128;
    __bf16* wt_ggr = bw;                     bw += 384 * 128;
    __bf16* wt_glr = bw;                     bw += 384 * 128;
    __bf16* wt_a1  = bw;                     bw += 128 * 128;
    __bf16* wt_a2  = bw;                     bw += 128 * 128;
    __bf16* wt_pcw0 = bw;                    bw += 256 * 256;
    __bf16* wt_pcw1 = bw;                    bw += 128 * 256;
    __bf16* wt_pcv0 = bw;                    bw += 256 * 128;
    __bf16* wt_pcv1 = bw;                    bw += 128 * 256;
    __bf16* wt_pvw0 = bw;                    bw += 256 * 256;
    __bf16* wt_pvw1 = bw;                    bw += 128 * 256;
    __bf16* wt_pvv0 = bw;                    bw += 256 * 128;
    __bf16* wt_pvv1 = bw;                    bw += 128 * 256;

    dim3 blk(256);
    // 0) pre-convert all weights to transposed bf16
    convert_wT_kernel<<<192, blk, 0, stream>>>(gg_k, wt_ggk, 128, 384);
    convert_wT_kernel<<<192, blk, 0, stream>>>(gl_k, wt_glk, 128, 384);
    convert_wT_kernel<<<192, blk, 0, stream>>>(gg_r, wt_ggr, 128, 384);
    convert_wT_kernel<<<192, blk, 0, stream>>>(gl_r, wt_glr, 128, 384);
    convert_wT_kernel<<<64,  blk, 0, stream>>>(a1w,  wt_a1,  128, 128);
    convert_wT_kernel<<<64,  blk, 0, stream>>>(a2w,  wt_a2,  128, 128);
    convert_wT_kernel<<<256, blk, 0, stream>>>(pc_w0, wt_pcw0, 256, 256);
    convert_wT_kernel<<<128, blk, 0, stream>>>(pc_w1, wt_pcw1, 256, 128);
    convert_wT_kernel<<<128, blk, 0, stream>>>(pc_v0, wt_pcv0, 128, 256);
    convert_wT_kernel<<<128, blk, 0, stream>>>(pc_v1, wt_pcv1, 256, 128);
    convert_wT_kernel<<<256, blk, 0, stream>>>(pv_w0, wt_pvw0, 256, 256);
    convert_wT_kernel<<<128, blk, 0, stream>>>(pv_w1, wt_pvw1, 256, 128);
    convert_wT_kernel<<<128, blk, 0, stream>>>(pv_v0, wt_pvv0, 128, 256);
    convert_wT_kernel<<<128, blk, 0, stream>>>(pv_v1, wt_pvv1, 256, 128);

    // 1) Input projections xz = x @ Wk + b[0]
    gemm_bf16_kernel<<<576, blk, 0, stream>>>(x, 128, wt_ggk, gg_b, xz_g, 384, 0, 128, 24, 0);
    gemm_bf16_kernel<<<576, blk, 0, stream>>>(x, 128, wt_glk, gl_b, xz_l, 384, 0, 128, 24, 0);

    // 2) Two GRU recurrences; h_g -> c_t[:, :128], h_l -> own buffer
    gru_kernel<<<2, 384, 0, stream>>>(xz_g, xz_l, wt_ggr, wt_glr, gg_b + 384, gl_b + 384,
                                      c_t, h_l, 256, 128);

    // 3) h1 = h_l @ a1, h2 = h_l @ a2
    gemm_bf16_kernel<<<192, blk, 0, stream>>>(h_l, 128, wt_a1, nullptr, h1, 128, 0, 128, 8, 0);
    gemm_bf16_kernel<<<192, blk, 0, stream>>>(h_l, 128, wt_a2, nullptr, h2, 128, 0, 128, 8, 0);

    // 4) causal additive attention -> c_t[:, 128:256]
    attn_kernel<<<Bn * 24, blk, 0, stream>>>(h1, h2, h_l, vv, c_t);

    // 5) head c
    gemm_bf16_kernel<<<384, blk, 0, stream>>>(c_t, 256, wt_pcw0, pc_b0, a0, 256, 0, 256, 16, 1);
    gemm_bf16_kernel<<<192, blk, 0, stream>>>(a0, 256, wt_pcw1, pc_b1, a1b, 128, 0, 256, 8, 1);
    gemm_bf16_kernel<<<384, blk, 0, stream>>>(x, 128, wt_pcv0, pc_c0, bb0, 256, 0, 128, 16, 1);
    gemm_bf16_kernel<<<192, blk, 0, stream>>>(bb0, 256, wt_pcv1, pc_c1, bb1, 128, 0, 256, 8, 1);
    head_dot_kernel<<<12, blk, 0, stream>>>(a1b, bb1, out, nullptr);

    // 6) head v (reuses buffers), multiplied by pred_c
    gemm_bf16_kernel<<<384, blk, 0, stream>>>(c_t, 256, wt_pvw0, pv_b0, a0, 256, 0, 256, 16, 1);
    gemm_bf16_kernel<<<192, blk, 0, stream>>>(a0, 256, wt_pvw1, pv_b1, a1b, 128, 0, 256, 8, 1);
    gemm_bf16_kernel<<<384, blk, 0, stream>>>(x, 128, wt_pvv0, pv_c0, bb0, 256, 0, 128, 16, 1);
    gemm_bf16_kernel<<<192, blk, 0, stream>>>(bb0, 256, wt_pvv1, pv_c1, bb1, 128, 0, 256, 8, 1);
    head_dot_kernel<<<12, blk, 0, stream>>>(a1b, bb1, out + ROWS, out);
}
